// SAttention_5497558139125
// MI455X (gfx1250) — compile-verified, matrix operand order checked by reading, not run on hardware
//
#include <hip/hip_runtime.h>
#include <hip/hip_bf16.h>

typedef _Float16 half8_t __attribute__((ext_vector_type(8)));
typedef _Float16 v16h    __attribute__((ext_vector_type(16)));
typedef float    v8f     __attribute__((ext_vector_type(8)));
typedef int      v4i     __attribute__((ext_vector_type(4)));

#define B_DIM 32
#define C_DIM 64
#define N_DIM 1024
#define T_DIM 64
#define QSTRIDE 72   // padded LDS row stride in halves (144B = 9*16B, keeps b128 alignment)

#if __has_builtin(__builtin_amdgcn_global_load_async_to_lds_b128) && \
    __has_builtin(__builtin_amdgcn_s_wait_asynccnt)
#define HAVE_ASYNC_LDS 1
#else
#define HAVE_ASYNC_LDS 0
#endif

static __device__ inline v16h combine16(half8_t lo, half8_t hi) {
    v16h r;
#pragma unroll
    for (int i = 0; i < 8; ++i) { r[i] = lo[i]; r[i + 8] = hi[i]; }
    return r;
}

static __device__ inline v8f wmma_f16(v16h a, v16h b, v8f c) {
    // D = A(16x32 f16) * B(32x16 f16) + C(16x16 f32)
    return __builtin_amdgcn_wmma_f32_16x16x32_f16(
        /*neg_a=*/false, a, /*neg_b=*/false, b,
        /*c_mod=*/(short)0, c, /*reuse_a=*/false, /*reuse_b=*/false);
}

// ---------------------------------------------------------------------------
// Kernel 1: q[b,n,t] = sum_c alpha[c] * x[b,c,n,t]   (memory-bound, 536 MB read)
// grid: B * N/4 blocks, 256 threads: 4 rows x 64 t per block, fully coalesced.
// ---------------------------------------------------------------------------
__global__ __launch_bounds__(256) void reduce_q_kernel(
    const float* __restrict__ x, const float* __restrict__ alpha,
    _Float16* __restrict__ qg) {
    __shared__ float a_s[C_DIM];
    const int tid = threadIdx.x;
    if (tid < C_DIM) a_s[tid] = alpha[tid];
    __syncthreads();

    const int b  = blockIdx.x >> 8;            // 256 n-blocks per batch
    const int n  = ((blockIdx.x & 255) << 2) + (tid >> 6);
    const int t  = tid & 63;

    const size_t cstride = (size_t)N_DIM * T_DIM;  // elements per channel plane
    const float* xp = x + (((size_t)b * C_DIM) * N_DIM + n) * T_DIM + t;

    float acc = 0.0f;
#pragma unroll 8
    for (int c = 0; c < C_DIM; ++c) {
        if (c + 4 < C_DIM)
            __builtin_prefetch(xp + (size_t)(c + 4) * cstride, 0, 0); // global_prefetch_b8
        acc = fmaf(a_s[c], xp[(size_t)c * cstride], acc);
    }
    qg[((size_t)b * N_DIM + n) * T_DIM + t] = (_Float16)acc;
}

// ---------------------------------------------------------------------------
// Kernel 2: fused  p = qW (f16 WMMA) ; S = p q^T (f16 WMMA, f32 acc) ; softmax
// grid: B * N/16 blocks (each owns 16 score rows), 256 threads = 8 waves.
// q chunks staged to LDS via gfx1250 async global->LDS copies when available.
// ---------------------------------------------------------------------------
__global__ __launch_bounds__(256) void scores_softmax_kernel(
    const _Float16* __restrict__ qg, const float* __restrict__ Wg,
    float* __restrict__ out) {
    __shared__ alignas(16) _Float16 q_lds[256 * QSTRIDE];  // 36 KB chunk of q rows
    __shared__ alignas(16) _Float16 p_lds[16 * QSTRIDE];   // p tile (16 x 64)
    __shared__ float red[8][16];

    const int tid  = threadIdx.x;
    const int lane = tid & 31;
    const int wv   = tid >> 5;
    const int b    = blockIdx.x >> 6;
    const int n0   = (blockIdx.x & 63) << 4;

    const int r15 = lane & 15;
    const int klo = (lane & 16) ? 8 : 0;   // 16-bit A/B K sub-block per lane half

    // ---- Phase 1: p[n0:n0+16, 0:64] = q_tile @ W, four s-tiles on waves 0..3
    if (wv < 4) {
        const int st = wv;
        const int s  = st * 16 + r15;                      // B column (lane 0-15 ids)
        const _Float16* qr = qg + ((size_t)b * N_DIM + n0 + r15) * T_DIM;
        v8f accp = {};
#pragma unroll
        for (int ks = 0; ks < 2; ++ks) {
            const int k0 = ks * 32;
            v16h A = combine16(*(const half8_t*)(qr + k0 + klo),
                               *(const half8_t*)(qr + k0 + 16 + klo));
            v16h Bv;
#pragma unroll
            for (int h = 0; h < 16; ++h) {
                const int kk = (h < 8) ? (klo + h) : (16 + klo + (h - 8));
                Bv[h] = (_Float16)Wg[(size_t)(k0 + kk) * T_DIM + s];
            }
            accp = wmma_f16(A, Bv, accp);
        }
#pragma unroll
        for (int j = 0; j < 8; ++j) {
            const int row = j + ((lane >> 4) << 3);        // C/D: vgpr j -> rows j, j+8
            p_lds[row * QSTRIDE + st * 16 + r15] = (_Float16)accp[j];
        }
    }
    __syncthreads();

    // ---- A operands (p rows) held in registers for the whole block
    v16h A0, A1;
    {
        const _Float16* pr = p_lds + r15 * QSTRIDE;
        A0 = combine16(*(const half8_t*)(pr + 0  + klo), *(const half8_t*)(pr + 16 + klo));
        A1 = combine16(*(const half8_t*)(pr + 32 + klo), *(const half8_t*)(pr + 48 + klo));
    }

    v8f acc[8];
#pragma unroll
    for (int i = 0; i < 8; ++i) acc[i] = (v8f){};

    // ---- Phase 2: loop over 4 chunks of 256 q-rows; 2 WMMAs per tile (K=64)
    for (int ci = 0; ci < 4; ++ci) {
        __syncthreads();   // previous chunk fully consumed
#if HAVE_ASYNC_LDS
        for (int i = tid; i < 256 * 8; i += 256) {         // 256 rows x 8 x b128
            const int row = i >> 3, seg = i & 7;
            __builtin_amdgcn_global_load_async_to_lds_b128(
                (v4i*)(qg + ((size_t)b * N_DIM + ci * 256 + row) * T_DIM + seg * 8),
                (v4i*)&q_lds[row * QSTRIDE + seg * 8],
                0, 0);
        }
        __builtin_amdgcn_s_wait_asynccnt(0);               // my LDS writes landed
        __syncthreads();                                   // everyone's landed
#else
        for (int i = tid; i < 256 * 8; i += 256) {         // 256 rows x 8 x b128
            const int row = i >> 3, seg = i & 7;
            *(uint4*)&q_lds[row * QSTRIDE + seg * 8] =
                *(const uint4*)(qg + ((size_t)b * N_DIM + ci * 256 + row) * T_DIM + seg * 8);
        }
        __syncthreads();
#endif
#pragma unroll
        for (int s = 0; s < 2; ++s) {
            const int mrow = (wv * 2 + s) * 16 + r15;      // B "column" = q row (q^T)
            const _Float16* qr = q_lds + mrow * QSTRIDE;
            v16h B0 = combine16(*(const half8_t*)(qr + 0  + klo), *(const half8_t*)(qr + 16 + klo));
            v16h B1 = combine16(*(const half8_t*)(qr + 32 + klo), *(const half8_t*)(qr + 48 + klo));
            const int ai = ci * 2 + s;
            acc[ai] = wmma_f16(A0, B0, acc[ai]);
            acc[ai] = wmma_f16(A1, B1, acc[ai]);
        }
    }

    // ---- Phase 3: softmax over the full 1024-wide rows
    float rmax[8], rsum[8];
#pragma unroll
    for (int j = 0; j < 8; ++j) {                          // per-row max (wave partial)
        float v = acc[0][j];
#pragma unroll
        for (int t = 1; t < 8; ++t) v = fmaxf(v, acc[t][j]);
#pragma unroll
        for (int m = 1; m <= 8; m <<= 1) v = fmaxf(v, __shfl_xor(v, m, 32));
        rmax[j] = v;
    }
    if (r15 == 0) {
#pragma unroll
        for (int j = 0; j < 8; ++j) red[wv][j + ((lane >> 4) << 3)] = rmax[j];
    }
    __syncthreads();
#pragma unroll
    for (int j = 0; j < 8; ++j) {
        const int row = j + ((lane >> 4) << 3);
        float v = red[0][row];
#pragma unroll
        for (int w = 1; w < 8; ++w) v = fmaxf(v, red[w][row]);
        rmax[j] = v;
    }
    __syncthreads();                                       // red reused for sums
#pragma unroll
    for (int j = 0; j < 8; ++j) {                          // exp + per-row sum
        float s = 0.0f;
#pragma unroll
        for (int t = 0; t < 8; ++t) {
            const float e = __expf(acc[t][j] - rmax[j]);
            acc[t][j] = e;
            s += e;
        }
#pragma unroll
        for (int m = 1; m <= 8; m <<= 1) s += __shfl_xor(s, m, 32);
        rsum[j] = s;
    }
    if (r15 == 0) {
#pragma unroll
        for (int j = 0; j < 8; ++j) red[wv][j + ((lane >> 4) << 3)] = rsum[j];
    }
    __syncthreads();
#pragma unroll
    for (int j = 0; j < 8; ++j) {
        const int row = j + ((lane >> 4) << 3);
        float s = 0.0f;
#pragma unroll
        for (int w = 0; w < 8; ++w) s += red[w][row];
        rsum[j] = 1.0f / s;
    }

    // ---- store: tile t covers cols [ (t>>1)*256 + (wv*2+(t&1))*16 , +16 )
#pragma unroll
    for (int t = 0; t < 8; ++t) {
        const int colb = (t >> 1) * 256 + (wv * 2 + (t & 1)) * 16;
        const int col  = colb + r15;
#pragma unroll
        for (int j = 0; j < 8; ++j) {
            const int row = j + ((lane >> 4) << 3);
            out[((size_t)b * N_DIM + n0 + row) * N_DIM + col] = acc[t][j] * rsum[j];
        }
    }
}

extern "C" void kernel_launch(void* const* d_in, const int* in_sizes, int n_in,
                              void* d_out, int out_size, void* d_ws, size_t ws_size,
                              hipStream_t stream) {
    const float* x     = (const float*)d_in[0];   // [32,64,1024,64]
    const float* W     = (const float*)d_in[1];   // [64,64]
    const float* alpha = (const float*)d_in[2];   // [64]
    float* out = (float*)d_out;                   // [32,1024,1024]

    _Float16* qg = (_Float16*)d_ws;               // 32*1024*64 f16 = 4 MB

    reduce_q_kernel<<<B_DIM * (N_DIM / 4), 256, 0, stream>>>(x, alpha, qg);
    scores_softmax_kernel<<<B_DIM * (N_DIM / 16), 256, 0, stream>>>(qg, W, out);
    (void)in_sizes; (void)n_in; (void)out_size; (void)ws_size;
}
